// MultiHeadAttention_14319420965065
// MI455X (gfx1250) — matrix operand order repaired and checked by reading, not compiled, not run on hardware
//
#include <hip/hip_runtime.h>
#include <hip/hip_bf16.h>
#include <math.h>

typedef __bf16 bf16_t;
typedef __attribute__((ext_vector_type(16))) __bf16 v16bf;
typedef __attribute__((ext_vector_type(8)))  __bf16 v8bf;
typedef __attribute__((ext_vector_type(8)))  float  v8f;
typedef __attribute__((ext_vector_type(4)))  float  v4f;
typedef __attribute__((ext_vector_type(4)))  unsigned int u32x4;
typedef __attribute__((ext_vector_type(8)))  int  i32x8;
typedef __attribute__((ext_vector_type(4)))  int  i32x4;

#define D_MODEL 1024
#define NHEADS  16
#define DK      64
#define SEQ     2048
#define BATCH   2
#define ROWS    (BATCH * SEQ)   // 4096
#define QKV_N   (3 * D_MODEL)   // 3072
#define KSTEPS  (D_MODEL / 32)  // 32

#if defined(__gfx1250__) && __has_builtin(__builtin_amdgcn_tensor_load_to_lds) && \
    __has_builtin(__builtin_amdgcn_s_wait_tensorcnt)
#define USE_TDM 1
#else
#define USE_TDM 0
#endif

// ---------------------------------------------------------------------------
// WMMA wrapper: D = A(16x32 bf16) x B(32x16 bf16) + C(16x16 f32)
// ---------------------------------------------------------------------------
static __device__ __forceinline__ v8f wmma_bf16(v16bf a, v16bf b, v8f c) {
  return __builtin_amdgcn_wmma_f32_16x16x32_bf16(
      false, a, false, b, (short)0, c, false, false);
}

static __device__ __forceinline__ v16bf cat8(v8bf lo, v8bf hi) {
  return __builtin_shufflevector(lo, hi, 0, 1, 2, 3, 4, 5, 6, 7,
                                         8, 9, 10, 11, 12, 13, 14, 15);
}

#if USE_TDM
// ---------------------------------------------------------------------------
// Tensor Data Mover: 2D bf16 tile (tileW elems x tileH rows, row stride
// strideElems) from global tile-start gbase into contiguous LDS at ldsOff.
// D# built per CDNA5 ISA ch.8: group0 = {count/type/addr}, group1 = dims.
// This toolchain's builtin takes 6 args (groups 0..3 + pad group + cpol).
// ---------------------------------------------------------------------------
static __device__ __forceinline__ void tdm_load_2d(const bf16_t* gbase,
                                                   unsigned ldsOff,
                                                   unsigned tileW, unsigned tileH,
                                                   unsigned strideElems) {
  const unsigned long long ga = (unsigned long long)(uintptr_t)gbase;
  u32x4 g0;
  g0[0] = 1u;                                        // count=1 (valid user D#)
  g0[1] = ldsOff;                                    // lds_addr (bytes)
  g0[2] = (unsigned)(ga & 0xFFFFFFFFu);              // global_addr[31:0]
  g0[3] = (unsigned)((ga >> 32) & 0x01FFFFFFu)       // global_addr[56:32]
          | 0x80000000u;                             // type=2 ("image")
  i32x8 g1;
  g1[0] = (int)(1u << 16);                           // data_size=1 -> 2 bytes
  g1[1] = (int)((tileW & 0xFFFFu) << 16);            // tensor_dim0[15:0]
  g1[2] = (int)((tileW >> 16) | ((tileH & 0xFFFFu) << 16));   // dim0 hi | dim1 lo
  g1[3] = (int)((tileH >> 16) | (tileW << 16));      // dim1 hi | tile_dim0
  g1[4] = (int)(tileH & 0xFFFFu);                    // tile_dim1 (tile_dim2=0)
  g1[5] = (int)strideElems;                          // tensor_dim0_stride[31:0]
  g1[6] = 0;                                         // stride hi, dim1_stride lo
  g1[7] = 0;
  const i32x4 z4 = {0, 0, 0, 0};
  const i32x8 z8 = {0, 0, 0, 0, 0, 0, 0, 0};
  __builtin_amdgcn_tensor_load_to_lds(g0, g1, z4, z4, z8, 0);
}
#endif

// ---------------------------------------------------------------------------
// Shared GEMM core: acc[4] (16x64 per wave) += A[128x1024] tile-row mBlock
// times W^T (W row-major [*,1024] bf16), columns nBlock..nBlock+63.
// A/B tiles staged in LDS; TDM double-buffered when available.
// ---------------------------------------------------------------------------
static __device__ __forceinline__ void gemm_core(const bf16_t* __restrict__ A,
                                                 const bf16_t* __restrict__ W,
                                                 int mBlock, int nBlock,
                                                 v8f acc[4]) {
  __shared__ __align__(64) bf16_t sA[2][128 * 32];   // 16 KB
  __shared__ __align__(64) bf16_t sB[2][64 * 32];    // 8 KB
  const int tid = threadIdx.x;
  const int wave = tid >> 5;
  const int lane = tid & 31;
  const int lane16 = lane & 15;
  const int hi = (lane >> 4) & 1;

#if USE_TDM
  if (wave == 0) {
    tdm_load_2d(A + (size_t)mBlock * D_MODEL + 0,  (unsigned)(uintptr_t)&sA[0][0], 32, 128, D_MODEL);
    tdm_load_2d(W + (size_t)nBlock * D_MODEL + 0,  (unsigned)(uintptr_t)&sB[0][0], 32, 64,  D_MODEL);
    tdm_load_2d(A + (size_t)mBlock * D_MODEL + 32, (unsigned)(uintptr_t)&sA[1][0], 32, 128, D_MODEL);
    tdm_load_2d(W + (size_t)nBlock * D_MODEL + 32, (unsigned)(uintptr_t)&sB[1][0], 32, 64,  D_MODEL);
  }
#endif

  for (int k = 0; k < KSTEPS; ++k) {
    const int buf = k & 1;
#if USE_TDM
    if (wave == 0) {
      if (k + 1 < KSTEPS) __builtin_amdgcn_s_wait_tensorcnt(2);  // stage k landed
      else                __builtin_amdgcn_s_wait_tensorcnt(0);
    }
    __syncthreads();
#else
    __syncthreads();
    const int k0 = k * 32;
    for (int idx = tid; idx < (128 * 32) / 8; idx += 256) {
      const int row = idx >> 2, part = idx & 3;
      *(v8bf*)&sA[buf][row * 32 + part * 8] =
          *(const v8bf*)(A + (size_t)(mBlock + row) * D_MODEL + k0 + part * 8);
    }
    for (int idx = tid; idx < (64 * 32) / 8; idx += 256) {
      const int row = idx >> 2, part = idx & 3;
      *(v8bf*)&sB[buf][row * 32 + part * 8] =
          *(const v8bf*)(W + (size_t)(nBlock + row) * D_MODEL + k0 + part * 8);
    }
    __syncthreads();
#endif

    // A fragment (16x32) from LDS: ds_load_b128 x2
    const bf16_t* pa = &sA[buf][(wave * 16 + lane16) * 32 + hi * 8];
    const v16bf a = cat8(*(const v8bf*)pa, *(const v8bf*)(pa + 16));
#pragma unroll
    for (int t = 0; t < 4; ++t) {
      const bf16_t* pb = &sB[buf][(t * 16 + lane16) * 32 + hi * 8];
      const v16bf b = cat8(*(const v8bf*)pb, *(const v8bf*)(pb + 16));
      acc[t] = wmma_bf16(a, b, acc[t]);
    }

#if USE_TDM
    __syncthreads();                                  // all waves done with buf
    if (wave == 0 && k + 2 < KSTEPS) {
      tdm_load_2d(A + (size_t)mBlock * D_MODEL + (k + 2) * 32,
                  (unsigned)(uintptr_t)&sA[buf][0], 32, 128, D_MODEL);
      tdm_load_2d(W + (size_t)nBlock * D_MODEL + (k + 2) * 32,
                  (unsigned)(uintptr_t)&sB[buf][0], 32, 64, D_MODEL);
    }
#endif
  }
}

// ---------------------------------------------------------------------------
// Elementwise f32 -> bf16 (8 elems/thread)
// ---------------------------------------------------------------------------
__global__ __launch_bounds__(256) void cvt_f32_bf16(const float* __restrict__ in,
                                                    bf16_t* __restrict__ out, int n8) {
  const int i = blockIdx.x * 256 + threadIdx.x;
  if (i < n8) {
    const v4f* p = (const v4f*)(in + (size_t)i * 8);
    const v4f a = p[0], b = p[1];
    v8bf o;
#pragma unroll
    for (int e = 0; e < 4; ++e) {
      o[e] = (__bf16)a[e];
      o[4 + e] = (__bf16)b[e];
    }
    *(v8bf*)(out + (size_t)i * 8) = o;
  }
}

// ---------------------------------------------------------------------------
// Kernel 1: fused QKV projection (bf16 inputs/weights).
// Scatters into Q (B,H,N,dk), K (B,H,N,dk), V transposed (B,H,dk,N).
// ---------------------------------------------------------------------------
__global__ __launch_bounds__(256) void qkv_gemm(const bf16_t* __restrict__ x,
                                                const bf16_t* __restrict__ w,
                                                const float* __restrict__ bias,
                                                bf16_t* __restrict__ Q,
                                                bf16_t* __restrict__ K,
                                                bf16_t* __restrict__ Vt) {
  const int wave = threadIdx.x >> 5;
  const int lane = threadIdx.x & 31;
  const int mBlock = blockIdx.y * 128;
  const int nBlock = blockIdx.x * 64;

  v8f acc[4] = {v8f{}, v8f{}, v8f{}, v8f{}};
  gemm_core(x, w, mBlock, nBlock, acc);

  const int lane16 = lane & 15;
  const int hi = (lane >> 4) & 1;
  const int mBase = mBlock + wave * 16;
#pragma unroll
  for (int t = 0; t < 4; ++t) {
    const int n = nBlock + t * 16 + lane16;
    const int region = n >> 10;                 // 0=Q 1=K 2=V
    const int f = n & (D_MODEL - 1);
    const int h = f >> 6;
    const int d = f & (DK - 1);
    const float bv = bias[n];
#pragma unroll
    for (int i = 0; i < 8; ++i) {
      const int r = mBase + i + hi * 8;         // global row
      const int bb = r >> 11;
      const int s = r & (SEQ - 1);
      const bf16_t val = (bf16_t)(acc[t][i] + bv);
      if (region == 0) {
        Q[(((size_t)(bb * NHEADS + h) * SEQ) + s) * DK + d] = val;
      } else if (region == 1) {
        K[(((size_t)(bb * NHEADS + h) * SEQ) + s) * DK + d] = val;
      } else {
        Vt[(((size_t)(bb * NHEADS + h) * DK) + d) * SEQ + s] = val;
      }
    }
  }
}

// ---------------------------------------------------------------------------
// bf16 fragment helpers for attention (layouts per ISA 7.12.2)
// ---------------------------------------------------------------------------
static __device__ __forceinline__ v16bf load_a_bf16(const bf16_t* __restrict__ base,
                                                    int ldr, int mBase, int k0, int lane) {
  const int lane16 = lane & 15;
  const int hi = (lane >> 4) & 1;
  const bf16_t* p = base + (size_t)(mBase + lane16) * ldr + k0 + hi * 8;
  return cat8(*(const v8bf*)p, *(const v8bf*)(p + 16));
}

static __device__ __forceinline__ v16bf load_bT_bf16(const bf16_t* __restrict__ W,
                                                     int ldr, int nBase, int k0, int lane) {
  const int lane16 = lane & 15;
  const int hi = (lane >> 4) & 1;
  const bf16_t* p = W + (size_t)(nBase + lane16) * ldr + k0 + hi * 8;
  return cat8(*(const v8bf*)p, *(const v8bf*)(p + 16));
}

// ---------------------------------------------------------------------------
// Kernel 2: flash attention. One wave per (b,h, 16-row Q tile).
// ---------------------------------------------------------------------------
__global__ __launch_bounds__(32) void attn_kernel(const bf16_t* __restrict__ Q,
                                                  const bf16_t* __restrict__ K,
                                                  const bf16_t* __restrict__ Vt,
                                                  bf16_t* __restrict__ O) {
  __shared__ __align__(16) bf16_t pbuf[16 * 32];

  const int lane = threadIdx.x & 31;
  const int lane16 = lane & 15;
  const int hi = (lane >> 4) & 1;

  const int rowTile = blockIdx.x & 127;
  const int bh = blockIdx.x >> 7;
  const int mBase = rowTile * 16;

  const bf16_t* Qb = Q + (size_t)bh * SEQ * DK;
  const bf16_t* Kb = K + (size_t)bh * SEQ * DK;
  const bf16_t* Vb = Vt + (size_t)bh * DK * SEQ;

  const v16bf aq0 = load_a_bf16(Qb, DK, mBase, 0, lane);
  const v16bf aq1 = load_a_bf16(Qb, DK, mBase, 32, lane);

  v8f o[4] = {v8f{}, v8f{}, v8f{}, v8f{}};
  float mrow[8], lrow[8];
#pragma unroll
  for (int i = 0; i < 8; ++i) { mrow[i] = -INFINITY; lrow[i] = 0.0f; }

  const float scale = 0.125f;  // 1/sqrt(64)

  for (int j0 = 0; j0 < SEQ; j0 += 32) {
    v8f s0 = v8f{}, s1 = v8f{};
    {
      v16bf b0 = load_bT_bf16(Kb, DK, j0, 0, lane);
      v16bf b1 = load_bT_bf16(Kb, DK, j0, 32, lane);
      s0 = wmma_bf16(aq0, b0, s0);
      s0 = wmma_bf16(aq1, b1, s0);
      v16bf b2 = load_bT_bf16(Kb, DK, j0 + 16, 0, lane);
      v16bf b3 = load_bT_bf16(Kb, DK, j0 + 16, 32, lane);
      s1 = wmma_bf16(aq0, b2, s1);
      s1 = wmma_bf16(aq1, b3, s1);
    }

    float alpha[8];
#pragma unroll
    for (int i = 0; i < 8; ++i) {
      float v0 = s0[i] * scale;
      float v1 = s1[i] * scale;
      float c = fmaxf(v0, v1);
#pragma unroll
      for (int m = 1; m < 16; m <<= 1) c = fmaxf(c, __shfl_xor(c, m, 16));
      const float mnew = fmaxf(mrow[i], c);
      alpha[i] = __expf(mrow[i] - mnew);
      mrow[i] = mnew;
      const float p0 = __expf(v0 - mnew);
      const float p1 = __expf(v1 - mnew);
      s0[i] = p0;
      s1[i] = p1;
      float rs = p0 + p1;
#pragma unroll
      for (int m = 1; m < 16; m <<= 1) rs += __shfl_xor(rs, m, 16);
      lrow[i] = lrow[i] * alpha[i] + rs;
    }

#pragma unroll
    for (int t = 0; t < 4; ++t)
#pragma unroll
      for (int i = 0; i < 8; ++i) o[t][i] *= alpha[i];

    // C-layout -> A-layout repack via LDS (same-wave DS ops are in-order)
#pragma unroll
    for (int i = 0; i < 8; ++i) {
      const int r = i + hi * 8;
      pbuf[r * 32 + lane16] = (bf16_t)s0[i];
      pbuf[r * 32 + 16 + lane16] = (bf16_t)s1[i];
    }
    v16bf ap;
    {
      const bf16_t* p = &pbuf[lane16 * 32 + hi * 8];
      ap = cat8(*(const v8bf*)p, *(const v8bf*)(p + 16));
    }

#pragma unroll
    for (int t = 0; t < 4; ++t) {
      v16bf bv = load_bT_bf16(Vb, SEQ, t * 16, j0, lane);
      o[t] = wmma_bf16(ap, bv, o[t]);
    }
  }

  const int b = bh >> 4;
  const int h = bh & 15;
#pragma unroll
  for (int t = 0; t < 4; ++t) {
    const int col = h * DK + t * 16 + lane16;
#pragma unroll
    for (int i = 0; i < 8; ++i) {
      const int row = b * SEQ + mBase + i + hi * 8;
      const float inv = 1.0f / lrow[i];
      O[(size_t)row * D_MODEL + col] = (bf16_t)(o[t][i] * inv);
    }
  }
}

// ---------------------------------------------------------------------------
// Kernel 3: output projection. out[4096,1024] = AO(bf16) @ out_w^T + out_b.
// ---------------------------------------------------------------------------
__global__ __launch_bounds__(256) void out_gemm(const bf16_t* __restrict__ A,
                                                const bf16_t* __restrict__ w,
                                                const float* __restrict__ bias,
                                                float* __restrict__ out) {
  const int wave = threadIdx.x >> 5;
  const int lane = threadIdx.x & 31;
  const int mBlock = blockIdx.y * 128;
  const int nBlock = blockIdx.x * 64;

  v8f acc[4] = {v8f{}, v8f{}, v8f{}, v8f{}};
  gemm_core(A, w, mBlock, nBlock, acc);

  const int lane16 = lane & 15;
  const int hi = (lane >> 4) & 1;
  const int mBase = mBlock + wave * 16;
#pragma unroll
  for (int t = 0; t < 4; ++t) {
    const int n = nBlock + t * 16 + lane16;
    const float bv = bias[n];
#pragma unroll
    for (int i = 0; i < 8; ++i) {
      const int r = mBase + i + hi * 8;
      out[(size_t)r * D_MODEL + n] = acc[t][i] + bv;
    }
  }
}

// ---------------------------------------------------------------------------
// Launch
// ---------------------------------------------------------------------------
extern "C" void kernel_launch(void* const* d_in, const int* in_sizes, int n_in,
                              void* d_out, int out_size, void* d_ws, size_t ws_size,
                              hipStream_t stream) {
  (void)in_sizes; (void)n_in; (void)out_size; (void)ws_size;
  const float* x     = (const float*)d_in[0];
  const float* qkv_w = (const float*)d_in[1];
  const float* qkv_b = (const float*)d_in[2];
  const float* out_w = (const float*)d_in[3];
  const float* out_b = (const float*)d_in[4];
  float* out = (float*)d_out;

  const size_t headElems = (size_t)BATCH * NHEADS * SEQ * DK;  // 4,194,304
  bf16_t* Q    = (bf16_t*)d_ws;
  bf16_t* K    = Q + headElems;
  bf16_t* Vt   = K + headElems;
  bf16_t* AO   = Vt + headElems;                                // ROWS*D_MODEL
  bf16_t* Xbf  = AO + (size_t)ROWS * D_MODEL;                   // ROWS*D_MODEL
  bf16_t* Wqkv = Xbf + (size_t)ROWS * D_MODEL;                  // QKV_N*D_MODEL
  bf16_t* Wout = Wqkv + (size_t)QKV_N * D_MODEL;                // D_MODEL^2
  // total workspace: 48 MB of bf16

  const int nx = ROWS * D_MODEL / 8;
  const int nw1 = QKV_N * D_MODEL / 8;
  const int nw2 = D_MODEL * D_MODEL / 8;
  cvt_f32_bf16<<<dim3((nx + 255) / 256),  dim3(256), 0, stream>>>(x, Xbf, nx);
  cvt_f32_bf16<<<dim3((nw1 + 255) / 256), dim3(256), 0, stream>>>(qkv_w, Wqkv, nw1);
  cvt_f32_bf16<<<dim3((nw2 + 255) / 256), dim3(256), 0, stream>>>(out_w, Wqkv == Wout ? Wout : Wout, nw2);

  qkv_gemm<<<dim3(QKV_N / 64, ROWS / 128), dim3(256), 0, stream>>>(
      Xbf, Wqkv, qkv_b, Q, K, Vt);
  attn_kernel<<<dim3(BATCH * NHEADS * (SEQ / 16)), dim3(32), 0, stream>>>(
      Q, K, Vt, AO);
  out_gemm<<<dim3(D_MODEL / 64, ROWS / 128), dim3(256), 0, stream>>>(
      AO, Wout, out_b, out);
}
